// ASPPup_78357383348639
// MI455X (gfx1250) — matrix operand order repaired and checked by reading, not compiled, not run on hardware
//
#include <hip/hip_runtime.h>
#include <stdint.h>

typedef __bf16 bf16;
typedef __attribute__((ext_vector_type(16))) __bf16 v16bf;
typedef __attribute__((ext_vector_type(8)))  __bf16 v8bf;
typedef __attribute__((ext_vector_type(8)))  float  v8f;
typedef __attribute__((ext_vector_type(4))) unsigned int u32x4;
typedef __attribute__((ext_vector_type(8))) int         i32x8;
typedef __attribute__((ext_vector_type(4))) int         i32x4;

#define HAVE_TDM __has_builtin(__builtin_amdgcn_tensor_load_to_lds)

__device__ __forceinline__ unsigned short bf_bits(bf16 h) {
    union { bf16 h; unsigned short u; } x; x.h = h; return x.u;
}

// Build a v16bf A-fragment from a row pointer (generic or LDS-derived).
// Contiguous K layout: VGPRs 0-3 <- K segment [16*(lane/16)..+8),
// VGPRs 4-7 <- [16*(lane/16) + 8..+8).
__device__ __forceinline__ v16bf load_a_frag(const bf16* __restrict__ row, int kBase, int sel) {
    v8bf lo = *(const v8bf*)(row + kBase + 16 * sel);
    v8bf hi = *(const v8bf*)(row + kBase + 16 * sel + 8);
    v16bf a;
#pragma unroll
    for (int i = 0; i < 8; ++i) { a[i] = lo[i]; a[i + 8] = hi[i]; }
    return a;
}

#if HAVE_TDM
// TDM: 2D tile load (rows x rowHalfs bf16, tile == tensor window) global -> LDS.
// 6-arg builtin variant (clang-23 / therock-10.0 headers).
// Descriptor packing per CDNA5 ISA sec 8.3/8.4:
//   group0: [1:0]=count=1, [63:32]=lds_addr, [120:64]=global_addr, [127:126]=type=2
//   group1: [17:16]=data_size(1=2B), [20]=pad_enable, [24:22]=pad_interval,
//           [31:25]=pad_amount, [79:48]=tensor_dim0, [111:80]=tensor_dim1,
//           [127:112]=tile_dim0, [143:128]=tile_dim1, [207:160]=dim0_stride,
//           [255:208]=dim1_stride
__device__ __forceinline__ void tdm_load_2d(unsigned lds_off, const bf16* gptr,
                                            unsigned rows, unsigned rowHalfs,
                                            unsigned w0pad) {
    unsigned long long ga = (unsigned long long)(size_t)gptr;
    u32x4 g0;
    g0[0] = 1u;                                             // count=1, load, user mode
    g0[1] = lds_off;                                        // lds_addr
    g0[2] = (unsigned)ga;                                   // global_addr[31:0]
    g0[3] = (unsigned)((ga >> 32) & 0x1FFFFFFu) | (2u << 30); // addr[56:32] | type=2
    i32x8 g1;
    g1[0] = (int)w0pad;                                     // mask=0, data_size, padding
    g1[1] = (int)((rowHalfs & 0xFFFFu) << 16);              // tensor_dim0[15:0]
    g1[2] = (int)(((rowHalfs >> 16) & 0xFFFFu) | ((rows & 0xFFFFu) << 16)); // dim0 hi | dim1 lo
    g1[3] = (int)(((rows >> 16) & 0xFFFFu) | ((rowHalfs & 0xFFFFu) << 16)); // dim1 hi | tile_dim0
    g1[4] = (int)(rows & 0xFFFFu);                          // tile_dim1 | tile_dim2=0
    g1[5] = (int)rowHalfs;                                  // dim0_stride[31:0]
    g1[6] = (int)((rowHalfs & 0xFFFFu) << 16);              // dim0_stride hi=0 | dim1_stride lo
    g1[7] = (int)((rowHalfs >> 16) & 0xFFFFu);              // dim1_stride hi
    i32x4 z4 = {0, 0, 0, 0};
    i32x8 z8 = {0, 0, 0, 0, 0, 0, 0, 0};
    __builtin_amdgcn_tensor_load_to_lds(g0, g1, z4, z4, z8, 0);
    __builtin_amdgcn_s_wait_tensorcnt((short)0);
}
#endif

// ---------------------------------------------------------------------------
// Pack x (8,256,64,64) fp32 NCHW -> bf16 channels-last [B,H,W,Cin]
// ---------------------------------------------------------------------------
__global__ __launch_bounds__(256) void k_pack_x(const float* __restrict__ x, bf16* __restrict__ xb) {
    __shared__ bf16 lds[32 * 33];
    const int ct = blockIdx.x & 7;
    const int st = blockIdx.x >> 3;
    const int cb = ct * 32;
    const int sBase = st * 32;
    const int b  = sBase >> 12;
    const int hw = sBase & 4095;
    const int h  = hw >> 6;
    const int w0 = hw & 63;
    const int t = threadIdx.x;
    const int r = t >> 5, col = t & 31;
#pragma unroll
    for (int i = 0; i < 4; ++i) {
        const int ch = cb + r + 8 * i;
        const float* src = x + ((((size_t)b * 256 + ch) * 64 + h) * 64 + w0 + col);
        __builtin_prefetch(src + 4096, 0, 0);
        lds[(r + 8 * i) * 33 + col] = (bf16)(*src);
    }
    __syncthreads();
    const int sp = t >> 3, q = t & 7;
    unsigned p0 = (unsigned)bf_bits(lds[(q * 4 + 0) * 33 + sp]) |
                  ((unsigned)bf_bits(lds[(q * 4 + 1) * 33 + sp]) << 16);
    unsigned p1 = (unsigned)bf_bits(lds[(q * 4 + 2) * 33 + sp]) |
                  ((unsigned)bf_bits(lds[(q * 4 + 3) * 33 + sp]) << 16);
    uint2 v; v.x = p0; v.y = p1;
    *(uint2*)(xb + (size_t)(sBase + sp) * 256 + cb + q * 4) = v;
}

__global__ __launch_bounds__(256) void k_pack_w3(const float* __restrict__ w, bf16* __restrict__ o, int n) {
    int idx = blockIdx.x * 256 + threadIdx.x;
    if (idx >= n) return;
    int ci  = idx & 255;
    int oc  = (idx >> 8) & 127;
    int tap = idx >> 15;
    o[idx] = (bf16)w[((size_t)(oc * 256 + ci)) * 9 + tap];
}

__global__ __launch_bounds__(256) void k_pack_dense(const float* __restrict__ w, bf16* __restrict__ o, int n) {
    int idx = blockIdx.x * 256 + threadIdx.x;
    if (idx < n) o[idx] = (bf16)w[idx];
}

__global__ void k_bn_prep(const float* __restrict__ g, const float* __restrict__ b,
                          const float* __restrict__ m, const float* __restrict__ v,
                          float* __restrict__ sc, float* __restrict__ sh) {
    int i = threadIdx.x;
    float s = g[i] * rsqrtf(v[i] + 1e-5f);
    sc[i] = s;
    sh[i] = b[i] - m[i] * s;
}

__global__ void k_zero(bf16* __restrict__ z) { z[blockIdx.x * 256 + threadIdx.x] = (bf16)0.f; }

// ---------------------------------------------------------------------------
// Branch conv as implicit GEMM. A (weights, per tap: 128x256 bf16) is staged
// to LDS by the Tensor Data Mover with pad_enable for a 264-half row pitch
// (bank-conflict-free, 16B aligned). B comes from L2-resident global memory.
// ---------------------------------------------------------------------------
#define APITCH 264   // 256 halfs + 4-DWORD TDM pad -> 528B row pitch
__global__ __launch_bounds__(256)
void k_branch_gemm(const bf16* __restrict__ xb, const bf16* __restrict__ Aw,
                   const float* __restrict__ scale, const float* __restrict__ shift,
                   const bf16* __restrict__ zbuf, bf16* __restrict__ up,
                   int taps, int dil, int br) {
    __shared__ bf16 smA[128 * APITCH];
    const int lane  = threadIdx.x & 31;
    const int wv    = threadIdx.x >> 5;
    const int Mbase = wv * 16;
    const int sel   = lane >> 4;
    const int lc    = lane & 15;
    const int colBase = blockIdx.x * 32;

    const int s0 = colBase + lc;
    const int s1 = s0 + 16;
    const int b0 = s0 >> 12, y0 = (s0 & 4095) >> 6, x0 = s0 & 63;
    const int b1 = s1 >> 12, y1 = (s1 & 4095) >> 6, x1 = s1 & 63;

    v8f acc0 = {}; v8f acc1 = {};

    for (int tap = 0; tap < taps; ++tap) {
        // Stage this tap's 128x256 weight tile into LDS via TDM (wave 0 only).
#if HAVE_TDM
        if (wv == 0) {
            // data_size=2B (bit16), pad_enable (bit20), interval=128DW (6<<22),
            // amount=4DW (3<<25)
            tdm_load_2d((unsigned)(size_t)smA, Aw + (size_t)tap * 128 * 256,
                        128u, 256u, (1u << 16) | (1u << 20) | (6u << 22) | (3u << 25));
        }
#else
        for (int i = threadIdx.x; i < 128 * 256; i += 256) {
            smA[(i >> 8) * APITCH + (i & 255)] = Aw[(size_t)tap * 128 * 256 + i];
        }
#endif
        __syncthreads();

        const int dy = (taps == 9) ? ((tap / 3) - 1) * dil : 0;
        const int dx = (taps == 9) ? ((tap % 3) - 1) * dil : 0;
        const int ty0 = y0 + dy, tx0 = x0 + dx;
        const int ty1 = y1 + dy, tx1 = x1 + dx;
        const bool ok0 = ((unsigned)ty0 < 64u) & ((unsigned)tx0 < 64u);
        const bool ok1 = ((unsigned)ty1 < 64u) & ((unsigned)tx1 < 64u);
        const bf16* bp0 = ok0 ? (xb + ((size_t)b0 * 4096 + ty0 * 64 + tx0) * 256) : zbuf;
        const bf16* bp1 = ok1 ? (xb + ((size_t)b1 * 4096 + ty1 * 64 + tx1) * 256) : zbuf;
        const bf16* arow = &smA[(Mbase + lc) * APITCH];
#pragma unroll
        for (int k = 0; k < 256; k += 32) {
            v16bf a  = load_a_frag(arow, k, sel);
            v16bf f0 = *(const v16bf*)(bp0 + k + 16 * sel);
            v16bf f1 = *(const v16bf*)(bp1 + k + 16 * sel);
            acc0 = __builtin_amdgcn_wmma_f32_16x16x32_bf16(false, a, false, f0, (short)0, acc0, false, false);
            acc1 = __builtin_amdgcn_wmma_f32_16x16x32_bf16(false, a, false, f1, (short)0, acc1, false, false);
        }
        __syncthreads();
    }

    const int ch0 = Mbase + 8 * sel;
    const int ry = br >> 1, rx = br & 1;
    {
        const int Y = 2 * y0 + ry, X = 2 * x0 + rx;
        size_t base = (((size_t)b0 * 128 + Y) * 128 + X) * 128 + ch0;
        unsigned pkk[4];
#pragma unroll
        for (int r = 0; r < 8; r += 2) {
            float f0 = fmaxf(fmaf(acc0[r],     scale[ch0 + r],     shift[ch0 + r]),     0.f);
            float f1 = fmaxf(fmaf(acc0[r + 1], scale[ch0 + r + 1], shift[ch0 + r + 1]), 0.f);
            pkk[r >> 1] = (unsigned)bf_bits((bf16)f0) | ((unsigned)bf_bits((bf16)f1) << 16);
        }
        uint4 pk; pk.x = pkk[0]; pk.y = pkk[1]; pk.z = pkk[2]; pk.w = pkk[3];
        *(uint4*)(up + base) = pk;
    }
    {
        const int Y = 2 * y1 + ry, X = 2 * x1 + rx;
        size_t base = (((size_t)b1 * 128 + Y) * 128 + X) * 128 + ch0;
        unsigned pkk[4];
#pragma unroll
        for (int r = 0; r < 8; r += 2) {
            float f0 = fmaxf(fmaf(acc1[r],     scale[ch0 + r],     shift[ch0 + r]),     0.f);
            float f1 = fmaxf(fmaf(acc1[r + 1], scale[ch0 + r + 1], shift[ch0 + r + 1]), 0.f);
            pkk[r >> 1] = (unsigned)bf_bits((bf16)f0) | ((unsigned)bf_bits((bf16)f1) << 16);
        }
        uint4 pk; pk.x = pkk[0]; pk.y = pkk[1]; pk.z = pkk[2]; pk.w = pkk[3];
        *(uint4*)(up + base) = pk;
    }
}

// ---------------------------------------------------------------------------
// Final 1x1 conv: K=128 GEMM, weights (128x128 bf16) staged to LDS by TDM
// with a 136-half padded pitch. BN+ReLU fused, fp32 NCHW output.
// ---------------------------------------------------------------------------
#define FPITCH 136   // 128 halfs + 4-DWORD TDM pad -> 272B row pitch
__global__ __launch_bounds__(256)
void k_final_gemm(const bf16* __restrict__ up, const bf16* __restrict__ Aw,
                  const float* __restrict__ scale, const float* __restrict__ shift,
                  float* __restrict__ out) {
    __shared__ bf16 smF[128 * FPITCH];
    const int lane  = threadIdx.x & 31;
    const int wv    = threadIdx.x >> 5;
    const int Mbase = wv * 16;
    const int sel   = lane >> 4;
    const int lc    = lane & 15;
    const int colBase = blockIdx.x * 32;

#if HAVE_TDM
    if (wv == 0) {
        // data_size=2B, pad_enable, interval=64DW (5<<22), amount=4DW (3<<25)
        tdm_load_2d((unsigned)(size_t)smF, Aw, 128u, 128u,
                    (1u << 16) | (1u << 20) | (5u << 22) | (3u << 25));
    }
#else
    for (int i = threadIdx.x; i < 128 * 128; i += 256) {
        smF[(i >> 7) * FPITCH + (i & 127)] = Aw[i];
    }
#endif
    __syncthreads();

    const int s0 = colBase + lc;
    const int s1 = s0 + 16;
    const int b0 = s0 >> 14, Y0 = (s0 & 16383) >> 7, X0 = s0 & 127;
    const int b1 = s1 >> 14, Y1 = (s1 & 16383) >> 7, X1 = s1 & 127;

    v8f acc0 = {}; v8f acc1 = {};
    const bf16* arow = &smF[(Mbase + lc) * FPITCH];
    const bf16* bp0  = up + (size_t)s0 * 128;
    const bf16* bp1  = up + (size_t)s1 * 128;
#pragma unroll
    for (int k = 0; k < 128; k += 32) {
        v16bf a  = load_a_frag(arow, k, sel);
        v16bf f0 = *(const v16bf*)(bp0 + k + 16 * sel);
        v16bf f1 = *(const v16bf*)(bp1 + k + 16 * sel);
        acc0 = __builtin_amdgcn_wmma_f32_16x16x32_bf16(false, a, false, f0, (short)0, acc0, false, false);
        acc1 = __builtin_amdgcn_wmma_f32_16x16x32_bf16(false, a, false, f1, (short)0, acc1, false, false);
    }

    const int ch0 = Mbase + 8 * sel;
#pragma unroll
    for (int r = 0; r < 8; ++r) {
        const int ch = ch0 + r;
        const float sc = scale[ch], sf = shift[ch];
        out[(((size_t)b0 * 128 + ch) * 128 + Y0) * 128 + X0] = fmaxf(fmaf(acc0[r], sc, sf), 0.f);
        out[(((size_t)b1 * 128 + ch) * 128 + Y1) * 128 + X1] = fmaxf(fmaf(acc1[r], sc, sf), 0.f);
    }
}

// ---------------------------------------------------------------------------
extern "C" void kernel_launch(void* const* d_in, const int* in_sizes, int n_in,
                              void* d_out, int out_size, void* d_ws, size_t ws_size,
                              hipStream_t stream) {
    (void)in_sizes; (void)n_in; (void)out_size; (void)ws_size;

    const float* x  = (const float*)d_in[0];
    const float* w0 = (const float*)d_in[1];
    const float* w1 = (const float*)d_in[2];
    const float* w2 = (const float*)d_in[3];
    const float* w3 = (const float*)d_in[4];
    const float* wp = (const float*)d_in[5];
    const float* bn[20];
    for (int i = 0; i < 20; ++i) bn[i] = (const float*)d_in[6 + i];

    char* p = (char*)d_ws;
    auto carve = [&](size_t bytes) -> char* {
        char* r = p; p += (bytes + 255) & ~(size_t)255; return r;
    };
    bf16* xb  = (bf16*)carve(8ull * 64 * 64 * 256 * 2);
    bf16* up  = (bf16*)carve(8ull * 128 * 128 * 128 * 2);
    bf16* aw1 = (bf16*)carve(9ull * 128 * 256 * 2);
    bf16* aw2 = (bf16*)carve(9ull * 128 * 256 * 2);
    bf16* aw3 = (bf16*)carve(9ull * 128 * 256 * 2);
    bf16* aw0 = (bf16*)carve(128ull * 256 * 2);
    bf16* awp = (bf16*)carve(128ull * 128 * 2);
    float* sc = (float*)carve(5ull * 128 * 4);
    float* sh = (float*)carve(5ull * 128 * 4);
    bf16* zb  = (bf16*)carve(1024);

    k_pack_x<<<8192, 256, 0, stream>>>(x, xb);
    k_pack_w3<<<(294912 + 255) / 256, 256, 0, stream>>>(w1, aw1, 294912);
    k_pack_w3<<<(294912 + 255) / 256, 256, 0, stream>>>(w2, aw2, 294912);
    k_pack_w3<<<(294912 + 255) / 256, 256, 0, stream>>>(w3, aw3, 294912);
    k_pack_dense<<<128, 256, 0, stream>>>(w0, aw0, 32768);
    k_pack_dense<<<64, 256, 0, stream>>>(wp, awp, 16384);
    for (int t = 0; t < 5; ++t)
        k_bn_prep<<<1, 128, 0, stream>>>(bn[4 * t + 0], bn[4 * t + 1], bn[4 * t + 2], bn[4 * t + 3],
                                         sc + 128 * t, sh + 128 * t);
    k_zero<<<2, 256, 0, stream>>>(zb);

    k_branch_gemm<<<1024, 256, 0, stream>>>(xb, aw0, sc + 0,   sh + 0,   zb, up, 1, 1,  0);
    k_branch_gemm<<<1024, 256, 0, stream>>>(xb, aw1, sc + 128, sh + 128, zb, up, 9, 6,  1);
    k_branch_gemm<<<1024, 256, 0, stream>>>(xb, aw2, sc + 256, sh + 256, zb, up, 9, 12, 2);
    k_branch_gemm<<<1024, 256, 0, stream>>>(xb, aw3, sc + 384, sh + 384, zb, up, 9, 18, 3);

    k_final_gemm<<<4096, 256, 0, stream>>>(up, awp, sc + 512, sh + 512, (float*)d_out);
}
